// PostFilterEnhancer_47880295415863
// MI455X (gfx1250) — compile-verified, matrix operand order checked by reading, not run on hardware
//
#include <hip/hip_runtime.h>
#include <math.h>

// ---------------------------------------------------------------------------
// Spectral-gating post filter on MI455X (gfx1250).
// STFT / iSTFT are expressed as f16 WMMA GEMMs (v_wmma_f32_16x16x32_f16)
// against a pre-packed windowed-DFT basis stored in B-fragment order.
// GEMM waves own 48x64 output strips (3 M-tiles x 4 N-tiles) so each
// B-fragment load feeds 3 WMMAs and each A-fragment feeds 4.
// ---------------------------------------------------------------------------

typedef _Float16 half_t;
typedef __attribute__((ext_vector_type(16))) _Float16 v16h;
typedef __attribute__((ext_vector_type(8)))  _Float16 v8h;
typedef __attribute__((ext_vector_type(8)))  float    v8f;

#define B_     16
#define L_     160000
#define T_     1251          // 1 + (160000+512-512)/128
#define M_     (B_ * T_)     // 20016 rows = 1251 tiles = 3 * 417
#define NFFT_  512
#define HOP_   128
#define F_     257
#define NF_PAD 576           // 2*257=514 fwd spec cols -> pad to mult of 64
#define KI_PAD 544           // inverse K (514) -> pad to mult of 32
#define FS_    272           // stride for mag/floor planes (17*16)
#define KT_F   16            // 512/32 k-tiles, forward
#define NT_F   36            // 576/16 n-tiles, forward
#define KT_I   17            // 544/32 k-tiles, inverse
#define NT_I   32            // 512/16 n-tiles, inverse
#define MG_    417           // 1251 / 3 M-tile groups
#define TWO_PI 6.28318530717958647692f

__device__ __forceinline__ float hann512(int n) {
    return 0.5f * (1.0f - __cosf(TWO_PI * (float)n * (1.0f / 512.0f)));
}

// --------------------------------------------------------------------------
// Build forward + inverse DFT bases as f16, pre-packed into WMMA B-fragment
// order: tile (kt,nt) is 512 halves, index = lane*16 + h, holding
// Bmat[k0 + kmap(h,hi)][n0 + (lane&15)] with kmap = (h&7) + 8*hi + 16*(h>>3).
// Window is folded into the basis; phase uses exact mod-512 reduction.
// --------------------------------------------------------------------------
__global__ void pf_build_basis(half_t* __restrict__ basF, half_t* __restrict__ basI) {
    int tid = blockIdx.x * blockDim.x + threadIdx.x;
    const int totF = KT_F * NT_F * 512;
    const int totI = KT_I * NT_I * 512;
    if (tid < totF) {
        int h = tid & 15, lane = (tid >> 4) & 31, tile = tid >> 9;
        int nt = tile % NT_F, kt = tile / NT_F;
        int hi = lane >> 4;
        int n  = nt * 16 + (lane & 15);                      // spec column
        int k  = kt * 32 + (h & 7) + 8 * hi + 16 * (h >> 3); // time sample
        float v = 0.0f;
        if (n < 2 * F_) {
            int   f  = n >> 1;
            int   fk = (f * k) & 511;                        // exact phase mod 512
            float th = TWO_PI * (float)fk * (1.0f / 512.0f);
            float w  = hann512(k);
            v = (n & 1) ? (-w * __sinf(th)) : (w * __cosf(th));
        }
        basF[tid] = (half_t)v;
    }
    if (tid < totI) {
        int h = tid & 15, lane = (tid >> 4) & 31, tile = tid >> 9;
        int nt = tile % NT_I, kt = tile / NT_I;
        int hi = lane >> 4;
        int n  = nt * 16 + (lane & 15);                      // time sample
        int k  = kt * 32 + (h & 7) + 8 * hi + 16 * (h >> 3); // spec column
        float v = 0.0f;
        if (k < 2 * F_) {
            int   f  = k >> 1;
            float c  = (f == 0 || f == 256) ? 1.0f : 2.0f;   // hermitian fold
            int   fn = (f * n) & 511;
            float th = TWO_PI * (float)fn * (1.0f / 512.0f);
            float w  = hann512(n) * c * (1.0f / 512.0f);     // synth window / N
            v = (k & 1) ? (-w * __sinf(th)) : (w * __cosf(th));
        }
        basI[tid] = (half_t)v;
    }
}

// --------------------------------------------------------------------------
// Extract reflect-padded frames as f16 (analysis window folded into basis).
// --------------------------------------------------------------------------
__global__ void pf_frames(const float* __restrict__ noisy,
                          const float* __restrict__ enh,
                          half_t* __restrict__ frN, half_t* __restrict__ frE) {
    long tid = (long)blockIdx.x * blockDim.x + threadIdx.x;
    if (tid >= (long)M_ * NFFT_) return;
    int  n = (int)(tid & 511);
    long r = tid >> 9;                 // b*T + t
    int  b = (int)(r / T_);
    int  t = (int)(r % T_);
    long s = (long)t * HOP_ + n - (NFFT_ / 2);
    if (s < 0)   s = -s;               // jnp reflect pad
    if (s >= L_) s = 2L * L_ - 2 - s;
    size_t src = (size_t)b * L_ + (size_t)s;
    frN[tid] = (half_t)noisy[src];
    frE[tid] = (half_t)enh[src];
}

// --------------------------------------------------------------------------
// WMMA GEMM: C(f32, M x Ntiles*16) = A(f16 row-major, lda) x Bpacked.
// One wave owns a 48x64 strip: 3 M-tiles x 4 N-tiles = 12 accumulators.
// Per k-step: 6 A b128 loads + 8 B b128 loads feed 12 WMMAs.
// --------------------------------------------------------------------------
__device__ __forceinline__ v16h pf_load_a_frag(const half_t* p, int hi) {
    v8h c0 = *(const v8h*)(p + 8 * hi);        // K = 8hi .. 8hi+7
    v8h c1 = *(const v8h*)(p + 16 + 8 * hi);   // K = 16+8hi .. 23+8hi
    return __builtin_shufflevector(c0, c1, 0, 1, 2, 3, 4, 5, 6, 7,
                                           8, 9, 10, 11, 12, 13, 14, 15);
}

__global__ __launch_bounds__(128)
void pf_wmma_gemm(const half_t* __restrict__ A, int lda,
                  const half_t* __restrict__ Bp,
                  float* __restrict__ C, int ldc,
                  int MtileGroups, int NtileGroups, int Ntiles, int Ktiles) {
    int wave = blockIdx.x * (blockDim.x >> 5) + (threadIdx.x >> 5);
    if (wave >= MtileGroups * NtileGroups) return;  // wave-uniform, EXEC stays all-1
    int mg = wave / NtileGroups;
    int ng = wave % NtileGroups;
    int lane = threadIdx.x & 31;
    int row  = lane & 15;
    int hi   = (lane >> 4) & 1;

    const half_t* Ar0 = A + (size_t)(mg * 48 + row) * lda;
    const half_t* Ar1 = Ar0 + (size_t)16 * lda;
    const half_t* Ar2 = Ar1 + (size_t)16 * lda;
    const half_t* Bbase = Bp + (size_t)(ng * 4) * 512 + lane * 16;

    v8f acc[3][4];
#pragma unroll
    for (int m = 0; m < 3; ++m)
#pragma unroll
        for (int n = 0; n < 4; ++n) acc[m][n] = (v8f){};

    for (int kt = 0; kt < Ktiles; ++kt) {
        int k0 = kt * 32;
        const half_t* Bt = Bbase + (size_t)kt * Ntiles * 512;
        v16h b0 = *(const v16h*)(Bt);
        v16h b1 = *(const v16h*)(Bt + 512);
        v16h b2 = *(const v16h*)(Bt + 1024);
        v16h b3 = *(const v16h*)(Bt + 1536);
        v16h a0 = pf_load_a_frag(Ar0 + k0, hi);
        v16h a1 = pf_load_a_frag(Ar1 + k0, hi);
        v16h a2 = pf_load_a_frag(Ar2 + k0, hi);

        acc[0][0] = __builtin_amdgcn_wmma_f32_16x16x32_f16(false, a0, false, b0, (short)0, acc[0][0], false, false);
        acc[0][1] = __builtin_amdgcn_wmma_f32_16x16x32_f16(false, a0, false, b1, (short)0, acc[0][1], false, false);
        acc[0][2] = __builtin_amdgcn_wmma_f32_16x16x32_f16(false, a0, false, b2, (short)0, acc[0][2], false, false);
        acc[0][3] = __builtin_amdgcn_wmma_f32_16x16x32_f16(false, a0, false, b3, (short)0, acc[0][3], false, false);
        acc[1][0] = __builtin_amdgcn_wmma_f32_16x16x32_f16(false, a1, false, b0, (short)0, acc[1][0], false, false);
        acc[1][1] = __builtin_amdgcn_wmma_f32_16x16x32_f16(false, a1, false, b1, (short)0, acc[1][1], false, false);
        acc[1][2] = __builtin_amdgcn_wmma_f32_16x16x32_f16(false, a1, false, b2, (short)0, acc[1][2], false, false);
        acc[1][3] = __builtin_amdgcn_wmma_f32_16x16x32_f16(false, a1, false, b3, (short)0, acc[1][3], false, false);
        acc[2][0] = __builtin_amdgcn_wmma_f32_16x16x32_f16(false, a2, false, b0, (short)0, acc[2][0], false, false);
        acc[2][1] = __builtin_amdgcn_wmma_f32_16x16x32_f16(false, a2, false, b1, (short)0, acc[2][1], false, false);
        acc[2][2] = __builtin_amdgcn_wmma_f32_16x16x32_f16(false, a2, false, b2, (short)0, acc[2][2], false, false);
        acc[2][3] = __builtin_amdgcn_wmma_f32_16x16x32_f16(false, a2, false, b3, (short)0, acc[2][3], false, false);
    }

#pragma unroll
    for (int m = 0; m < 3; ++m) {
        float* Cbase = C + (size_t)(mg * 48 + m * 16) * ldc + ng * 64 + (lane & 15);
#pragma unroll
        for (int i = 0; i < 8; ++i) {
            size_t roff = (size_t)(i + 8 * hi) * ldc;
            Cbase[roff + 0]  = acc[m][0][i];
            Cbase[roff + 16] = acc[m][1][i];
            Cbase[roff + 32] = acc[m][2][i];
            Cbase[roff + 48] = acc[m][3][i];
        }
    }
}

// -------------------- elementwise DSP stages ------------------------------
__global__ void pf_mag(const float* __restrict__ spec, float* __restrict__ mag) {
    long tid = (long)blockIdx.x * blockDim.x + threadIdx.x;
    if (tid >= (long)M_ * F_) return;
    int    f = (int)(tid % F_);
    size_t r = (size_t)(tid / F_);
    float re = spec[r * NF_PAD + 2 * f];
    float im = spec[r * NF_PAD + 2 * f + 1];
    mag[r * FS_ + f] = fmaxf(__fsqrt_rn(re * re + im * im), 1e-6f);
}

// rolling q=0.1 quantile over 31 frames == 4th-smallest (edge/replicate pad)
__global__ void pf_quantile(const float* __restrict__ mag, float* __restrict__ fl) {
    long tid = (long)blockIdx.x * blockDim.x + threadIdx.x;
    if (tid >= (long)M_ * F_) return;
    int    f = (int)(tid % F_);
    size_t r = (size_t)(tid / F_);
    int    b = (int)(r / T_), t = (int)(r % T_);
    size_t base = (size_t)b * T_;
    float m0 = 3.4e38f, m1 = 3.4e38f, m2 = 3.4e38f, m3 = 3.4e38f;
    for (int i = -15; i <= 15; ++i) {
        int tc = t + i;
        tc = tc < 0 ? 0 : (tc > T_ - 1 ? T_ - 1 : tc);
        float v = mag[(base + tc) * FS_ + f];
        if (v < m3) {
            if (v < m2) { m3 = m2;
                if (v < m1) { m2 = m1;
                    if (v < m0) { m1 = m0; m0 = v; } else m1 = v;
                } else m2 = v;
            } else m3 = v;
        }
    }
    fl[r * FS_ + f] = m3;
}

__global__ void pf_pool_freq(const float* __restrict__ in, float* __restrict__ out) {
    long tid = (long)blockIdx.x * blockDim.x + threadIdx.x;
    if (tid >= (long)M_ * F_) return;
    int    f = (int)(tid % F_);
    size_t r = (size_t)(tid / F_);
    float s = 0.0f;
    for (int d = -2; d <= 2; ++d) {
        int ff = f + d;
        if (ff >= 0 && ff < F_) s += in[r * FS_ + ff];   // zero pad
    }
    out[r * FS_ + f] = s * (1.0f / 5.0f);
}

__global__ void pf_pool_time(const float* __restrict__ in, float* __restrict__ out) {
    long tid = (long)blockIdx.x * blockDim.x + threadIdx.x;
    if (tid >= (long)M_ * F_) return;
    int    f = (int)(tid % F_);
    size_t r = (size_t)(tid / F_);
    int    b = (int)(r / T_), t = (int)(r % T_);
    size_t base = (size_t)b * T_;
    float s = 0.0f;
    for (int d = -4; d <= 4; ++d) {
        int tc = t + d;
        if (tc >= 0 && tc < T_) s += in[(base + tc) * FS_ + f];   // zero pad
    }
    out[r * FS_ + f] = fmaxf(s * (1.0f / 9.0f), 1e-6f);
}

// sigmoid gate + unit-phase re-synthesis; emit f16 A-matrix for inverse GEMM
__global__ void pf_mask_gate(const float* __restrict__ espec,
                             const float* __restrict__ floor3,
                             half_t* __restrict__ gated) {
    long tid = (long)blockIdx.x * blockDim.x + threadIdx.x;
    if (tid >= (long)M_ * (NF_PAD / 2)) return;
    int    j = (int)(tid % (NF_PAD / 2));
    size_t r = (size_t)(tid / (NF_PAD / 2));
    if (j >= F_) {                                    // zero the K padding
        gated[r * NF_PAD + 2 * j]     = (half_t)0.0f;
        gated[r * NF_PAD + 2 * j + 1] = (half_t)0.0f;
        return;
    }
    float re  = espec[r * NF_PAD + 2 * j];
    float im  = espec[r * NF_PAD + 2 * j + 1];
    float mgr = __fsqrt_rn(re * re + im * im);
    float emg = fmaxf(mgr, 1e-6f);
    float fl  = floor3[r * FS_ + j];
    float x   = (emg - 1.5f * fl) / (0.15f * fl + 1e-6f);
    float mk  = 1.0f / (1.0f + __expf(-x));
    float fm  = 0.08f + 0.92f * (0.65f + 0.35f * mk);
    float sc  = emg * fm / fmaxf(mgr, 1e-12f);        // |e|*mask * unit phase
    gated[r * NF_PAD + 2 * j]     = (half_t)(re * sc);
    gated[r * NF_PAD + 2 * j + 1] = (half_t)(im * sc);
}

// overlap-add with window^2 normalization; trim center pad
__global__ void pf_overlap_add(const float* __restrict__ fr, float* __restrict__ out) {
    long tid = (long)blockIdx.x * blockDim.x + threadIdx.x;
    if (tid >= (long)B_ * L_) return;
    int p = (int)(tid % L_);
    int b = (int)(tid / L_);
    int q = p + NFFT_ / 2;
    int tmin = q - (NFFT_ - 1);
    tmin = tmin < 0 ? 0 : ((tmin + HOP_ - 1) >> 7);
    int tmax = q >> 7; if (tmax > T_ - 1) tmax = T_ - 1;
    float s = 0.0f, wsq = 0.0f;
    for (int t = tmin; t <= tmax; ++t) {
        int n = q - (t << 7);
        s += fr[((size_t)b * T_ + t) * NFFT_ + n];
        float w = hann512(n);
        wsq += w * w;
    }
    out[tid] = s / fmaxf(wsq, 1e-11f);
}

// --------------------------------------------------------------------------
extern "C" void kernel_launch(void* const* d_in, const int* in_sizes, int n_in,
                              void* d_out, int out_size, void* d_ws, size_t ws_size,
                              hipStream_t stream) {
    (void)in_sizes; (void)n_in; (void)out_size; (void)ws_size;
    const float* noisy = (const float*)d_in[0];
    const float* enh   = (const float*)d_in[1];
    float* out = (float*)d_out;
    char*  w   = (char*)d_ws;

    // workspace layout (aliased regions, strict producer->consumer order)
    size_t o = 0;
    auto take = [&](size_t n) { size_t r = o; o += (n + 255) & ~(size_t)255; return r; };
    half_t* basF = (half_t*)(w + take((size_t)KT_F * NT_F * 512 * 2)); // 0.59 MB
    half_t* basI = (half_t*)(w + take((size_t)KT_I * NT_I * 512 * 2)); // 0.56 MB
    char* regA = w + take((size_t)M_ * FS_    * 4);  // frames_e -> n_mag -> floor3
    char* regB = w + take((size_t)M_ * NF_PAD * 2);  // frames_n -> floor1 -> gated
    char* regC = w + take((size_t)M_ * NF_PAD * 4);  // e_spec -> frames_out
    char* regD = w + take((size_t)M_ * NF_PAD * 4);  // n_spec -> floor2

    half_t* frames_e = (half_t*)regA;  float* n_mag  = (float*)regA;  float* floor3 = (float*)regA;
    half_t* frames_n = (half_t*)regB;  float* floor1 = (float*)regB;  half_t* gated = (half_t*)regB;
    float*  e_spec   = (float*)regC;   float* frames_out = (float*)regC;
    float*  n_spec   = (float*)regD;   float* floor2 = (float*)regD;

    const int THR = 256;
    auto blocks = [](long total, int thr) { return (unsigned)((total + thr - 1) / thr); };

    // 1) bases (packed for WMMA B fragments)
    pf_build_basis<<<blocks((long)KT_F * NT_F * 512, THR), THR, 0, stream>>>(basF, basI);

    // 2) f16 frame matrices (reflect pad)
    pf_frames<<<blocks((long)M_ * NFFT_, THR), THR, 0, stream>>>(noisy, enh, frames_n, frames_e);

    // 3) forward STFT GEMMs: (20016 x 512) x (512 x 576), 48x64 per wave
    {
        int ngr = NT_F / 4;
        unsigned g = (unsigned)((MG_ * ngr + 3) / 4);
        pf_wmma_gemm<<<g, 128, 0, stream>>>(frames_e, NFFT_, basF, e_spec, NF_PAD, MG_, ngr, NT_F, KT_F);
        pf_wmma_gemm<<<g, 128, 0, stream>>>(frames_n, NFFT_, basF, n_spec, NF_PAD, MG_, ngr, NT_F, KT_F);
    }

    // 4) noisy magnitude, rolling 4th-smallest floor, pooling
    pf_mag      <<<blocks((long)M_ * F_, THR), THR, 0, stream>>>(n_spec, n_mag);
    pf_quantile <<<blocks((long)M_ * F_, THR), THR, 0, stream>>>(n_mag, floor1);
    pf_pool_freq<<<blocks((long)M_ * F_, THR), THR, 0, stream>>>(floor1, floor2);
    pf_pool_time<<<blocks((long)M_ * F_, THR), THR, 0, stream>>>(floor2, floor3);

    // 5) sigmoid gate -> gated spectrum (f16)
    pf_mask_gate<<<blocks((long)M_ * (NF_PAD / 2), THR), THR, 0, stream>>>(e_spec, floor3, gated);

    // 6) inverse STFT GEMM: (20016 x 544) x (544 x 512), 48x64 per wave
    {
        int ngr = NT_I / 4;
        unsigned g = (unsigned)((MG_ * ngr + 3) / 4);
        pf_wmma_gemm<<<g, 128, 0, stream>>>(gated, NF_PAD, basI, frames_out, NFFT_, MG_, ngr, NT_I, KT_I);
    }

    // 7) overlap-add + wsq normalization, trim pad
    pf_overlap_add<<<blocks((long)B_ * L_, THR), THR, 0, stream>>>(frames_out, out);
}
